// Attention_996432413012
// MI455X (gfx1250) — matrix operand order repaired and checked by reading, not compiled, not run on hardware
//
#include <hip/hip_runtime.h>
#include <hip/hip_bf16.h>

typedef __attribute__((ext_vector_type(16))) _Float16 v16h;
typedef __attribute__((ext_vector_type(8)))  float    v8f;

// Problem dims
#define RR   64
#define CC   512
#define EE   768
#define HH   12
#define DD   64
#define RD   4096   // RR*DD
#define MM   32768  // RR*CC (B=1)
#define OUT_ATTN_OFF  25165824   // MM*EE
#define ATTN_ELEMS    3145728    // HH*CC*CC

// ---------------------------------------------------------------------------
// Fragment loaders matching CDNA5 WMMA VGPR layouts (16-bit, 16x16x32)
// ---------------------------------------------------------------------------

// A-matrix 16x32 (MxK), row-major source with leading dim lda.
// lanes 0-15: M=lane, K in {0..7} (elems 0..7) and {16..23} (elems 8..15)
// lanes 16-31: M=lane-16, K in {8..15} and {24..31}
__device__ __forceinline__ v16h load_a_frag(const _Float16* __restrict__ A, int lda) {
  int lane = threadIdx.x & 31;
  int g    = lane >> 4;
  int m    = lane & 15;
  const _Float16* row = A + (size_t)m * lda;
  int k0 = g * 8;
  int k1 = 16 + g * 8;
  v16h a;
#pragma unroll
  for (int i = 0; i < 8; ++i) a[i] = row[k0 + i];
#pragma unroll
  for (int i = 0; i < 8; ++i) a[8 + i] = row[k1 + i];
  return a;
}

// B-matrix 32x16 (KxN) where the source holds B^T row-major: W[n, k], ld = ldw.
// lanes 0-15: N=lane, K in {0..7} and {16..23}; lanes 16-31: N=lane-16, K in {8..15} and {24..31}.
__device__ __forceinline__ v16h load_b_frag(const _Float16* __restrict__ W, int ldw) {
  int lane = threadIdx.x & 31;
  int g    = lane >> 4;
  int n    = lane & 15;
  const _Float16* row = W + (size_t)n * ldw + g * 8;
  v16h b;
#pragma unroll
  for (int i = 0; i < 16; ++i) b[i] = row[i < 8 ? i : i + 8];
  return b;
}

// Wave-level GEMM core: C[64x64] += A[64xK] * W[64xK]^T, K multiple of 32.
// 4x4 accumulator tiles: 16 WMMAs per 32-wide K step from 8 fragment loads.
__device__ __forceinline__ void gemm_wave(const _Float16* __restrict__ A, int lda,
                                          const _Float16* __restrict__ W, int ldw,
                                          int K, v8f c[4][4]) {
  for (int k = 0; k < K; k += 32) {
    v16h a[4];
    v16h b[4];
#pragma unroll
    for (int t = 0; t < 4; ++t) a[t] = load_a_frag(A + (size_t)(t * 16) * lda + k, lda);
#pragma unroll
    for (int t = 0; t < 4; ++t) b[t] = load_b_frag(W + (size_t)(t * 16) * ldw + k, ldw);
#pragma unroll
    for (int i = 0; i < 4; ++i)
#pragma unroll
      for (int j = 0; j < 4; ++j)
        c[i][j] = __builtin_amdgcn_wmma_f32_16x16x32_f16(
            false, a[i], false, b[j], (short)0, c[i][j], false, false);
  }
}

// Block = 256 threads = 8 waves tiled 2(M) x 4(N); wave tile 64x64;
// block tile 128(M) x 256(N).
__device__ __forceinline__ void tile_coords(int& m0, int& n0) {
  int wave = threadIdx.x >> 5;
  int wm = wave >> 2;   // 0..1
  int wn = wave & 3;    // 0..3
  m0 = blockIdx.y * 128 + wm * 64;
  n0 = blockIdx.x * 256 + wn * 64;
}

// ---------------------------------------------------------------------------
// f32 -> f16 conversion
// ---------------------------------------------------------------------------
__global__ __launch_bounds__(256) void cvt_f32_f16(const float* __restrict__ s,
                                                   _Float16* __restrict__ d, int n) {
  int idx = blockIdx.x * blockDim.x + threadIdx.x;
  int stride = gridDim.x * blockDim.x;
  for (int i = idx; i < n; i += stride) d[i] = (_Float16)s[i];
}

// ---------------------------------------------------------------------------
// QKV projection: y = x @ W^T + b, scattered into attention layouts.
// MODE 0: q16[h][i][r*D+d] *= 1/64   (scaling = D^-0.5 / sqrt(R))
// MODE 1: k16[h][j][r*D+d]
// MODE 2: v16[h][r*D+d][j]  (transposed so context GEMM B-loads are contiguous)
// ---------------------------------------------------------------------------
template <int MODE>
__global__ __launch_bounds__(256) void proj_qkv(const _Float16* __restrict__ x16,
                                                const _Float16* __restrict__ w16,
                                                const float* __restrict__ bias,
                                                _Float16* __restrict__ out16) {
  int m0, n0;
  tile_coords(m0, n0);
  v8f c[4][4];
#pragma unroll
  for (int i = 0; i < 4; ++i)
#pragma unroll
    for (int j = 0; j < 4; ++j) c[i][j] = (v8f)(0.0f);

  gemm_wave(x16 + (size_t)m0 * EE, EE, w16 + (size_t)n0 * EE, EE, EE, c);

  int lane = threadIdx.x & 31;
  int g = lane >> 4;
  int nn = lane & 15;
#pragma unroll
  for (int i = 0; i < 4; ++i)
#pragma unroll
    for (int j = 0; j < 4; ++j)
#pragma unroll
      for (int r8 = 0; r8 < 8; ++r8) {
        int row = m0 + i * 16 + g * 8 + r8;   // m = r*CC + ci
        int col = n0 + j * 16 + nn;           // e = h*DD + d
        float val = c[i][j][r8] + bias[col];
        int r  = row >> 9;
        int ci = row & (CC - 1);
        int h  = col >> 6;
        int d  = col & (DD - 1);
        if (MODE == 0) {
          out16[((size_t)h * CC + ci) * RD + r * DD + d] = (_Float16)(val * 0.015625f);
        } else if (MODE == 1) {
          out16[((size_t)h * CC + ci) * RD + r * DD + d] = (_Float16)val;
        } else {
          out16[((size_t)h * RD + (r * DD + d)) * CC + ci] = (_Float16)val;
        }
      }
}

// ---------------------------------------------------------------------------
// Logits: per head attn[i,j] = sum_{r,d} q[i,rd] * k[j,rd]; K = 4096, f32 out.
// ---------------------------------------------------------------------------
__global__ __launch_bounds__(256) void logits_kernel(const _Float16* __restrict__ q16,
                                                     const _Float16* __restrict__ k16,
                                                     float* __restrict__ attn) {
  int h = blockIdx.z;
  int m0, n0;
  tile_coords(m0, n0);
  const _Float16* qh = q16 + (size_t)h * CC * RD;
  const _Float16* kh = k16 + (size_t)h * CC * RD;
  v8f c[4][4];
#pragma unroll
  for (int i = 0; i < 4; ++i)
#pragma unroll
    for (int j = 0; j < 4; ++j) c[i][j] = (v8f)(0.0f);

  gemm_wave(qh + (size_t)m0 * RD, RD, kh + (size_t)n0 * RD, RD, RD, c);

  float* ah = attn + (size_t)h * CC * CC;
  int lane = threadIdx.x & 31;
  int g = lane >> 4;
  int nn = lane & 15;
#pragma unroll
  for (int i = 0; i < 4; ++i)
#pragma unroll
    for (int j = 0; j < 4; ++j)
#pragma unroll
      for (int r8 = 0; r8 < 8; ++r8) {
        int row = m0 + i * 16 + g * 8 + r8;
        int col = n0 + j * 16 + nn;
        ah[(size_t)row * CC + col] = c[i][j][r8];
      }
}

// ---------------------------------------------------------------------------
// Softmax over j with diagonal masked; in-place f32 + f16 copy for next GEMM.
// One 256-thread block per (h, i) row of 512.
// ---------------------------------------------------------------------------
__global__ __launch_bounds__(256) void softmax_kernel(float* __restrict__ attn,
                                                      _Float16* __restrict__ p16) {
  __shared__ float red[256];
  int row = blockIdx.x;        // h*CC + i
  int i   = row & (CC - 1);
  float* L = attn + (size_t)row * CC;
  int t = threadIdx.x;

  float v0 = L[t];
  float v1 = L[t + 256];
  if (t == i) v0 = -1e9f;
  if (t + 256 == i) v1 = -1e9f;

  red[t] = fmaxf(v0, v1);
  __syncthreads();
  for (int s = 128; s > 0; s >>= 1) {
    if (t < s) red[t] = fmaxf(red[t], red[t + s]);
    __syncthreads();
  }
  float mx = red[0];
  __syncthreads();

  float e0 = __expf(v0 - mx);
  float e1 = __expf(v1 - mx);
  red[t] = e0 + e1;
  __syncthreads();
  for (int s = 128; s > 0; s >>= 1) {
    if (t < s) red[t] += red[t + s];
    __syncthreads();
  }
  float inv = 1.0f / red[0];

  float p0 = e0 * inv;
  float p1 = e1 * inv;
  L[t] = p0;
  L[t + 256] = p1;
  _Float16* P = p16 + (size_t)row * CC;
  P[t] = (_Float16)p0;
  P[t + 256] = (_Float16)p1;
}

// ---------------------------------------------------------------------------
// Context: per head ctx[i, r*D+d] = sum_j P[i,j] * vT[r*D+d, j]; K = 512.
// Scatter back to [m = r*CC+i, e = h*DD+d] f16 for the output projection.
// ---------------------------------------------------------------------------
__global__ __launch_bounds__(256) void context_kernel(const _Float16* __restrict__ p16,
                                                      const _Float16* __restrict__ v16,
                                                      _Float16* __restrict__ ctx16) {
  int h = blockIdx.z;
  int m0, n0;
  tile_coords(m0, n0);
  const _Float16* ph = p16 + (size_t)h * CC * CC;
  const _Float16* vh = v16 + (size_t)h * RD * CC;
  v8f c[4][4];
#pragma unroll
  for (int i = 0; i < 4; ++i)
#pragma unroll
    for (int j = 0; j < 4; ++j) c[i][j] = (v8f)(0.0f);

  gemm_wave(ph + (size_t)m0 * CC, CC, vh + (size_t)n0 * CC, CC, CC, c);

  int lane = threadIdx.x & 31;
  int g = lane >> 4;
  int nn = lane & 15;
#pragma unroll
  for (int i = 0; i < 4; ++i)
#pragma unroll
    for (int j = 0; j < 4; ++j)
#pragma unroll
      for (int r8 = 0; r8 < 8; ++r8) {
        int row = m0 + i * 16 + g * 8 + r8;   // i in [0, CC)
        int col = n0 + j * 16 + nn;           // r*DD + d
        int r = col >> 6;
        int d = col & (DD - 1);
        ctx16[((size_t)r * CC + row) * EE + h * DD + d] = (_Float16)c[i][j][r8];
      }
}

// ---------------------------------------------------------------------------
// Output projection: out = ctx @ Wo^T + bo, f32 to d_out.
// ---------------------------------------------------------------------------
__global__ __launch_bounds__(256) void outproj_kernel(const _Float16* __restrict__ ctx16,
                                                      const _Float16* __restrict__ wo16,
                                                      const float* __restrict__ bo,
                                                      float* __restrict__ out) {
  int m0, n0;
  tile_coords(m0, n0);
  v8f c[4][4];
#pragma unroll
  for (int i = 0; i < 4; ++i)
#pragma unroll
    for (int j = 0; j < 4; ++j) c[i][j] = (v8f)(0.0f);

  gemm_wave(ctx16 + (size_t)m0 * EE, EE, wo16 + (size_t)n0 * EE, EE, EE, c);

  int lane = threadIdx.x & 31;
  int g = lane >> 4;
  int nn = lane & 15;
#pragma unroll
  for (int i = 0; i < 4; ++i)
#pragma unroll
    for (int j = 0; j < 4; ++j)
#pragma unroll
      for (int r8 = 0; r8 < 8; ++r8) {
        int row = m0 + i * 16 + g * 8 + r8;
        int col = n0 + j * 16 + nn;
        out[(size_t)row * EE + col] = c[i][j][r8] + bo[col];
      }
}

// ---------------------------------------------------------------------------
extern "C" void kernel_launch(void* const* d_in, const int* in_sizes, int n_in,
                              void* d_out, int out_size, void* d_ws, size_t ws_size,
                              hipStream_t stream) {
  const float* x  = (const float*)d_in[0];
  // d_in[1] = network (unused by reference)
  const float* Wq = (const float*)d_in[2];
  const float* bq = (const float*)d_in[3];
  const float* Wk = (const float*)d_in[4];
  const float* bk = (const float*)d_in[5];
  const float* Wv = (const float*)d_in[6];
  const float* bv = (const float*)d_in[7];
  const float* Wo = (const float*)d_in[8];
  const float* bo = (const float*)d_in[9];
  const float* l  = (const float*)d_in[10];

  float* out     = (float*)d_out;
  float* attnOut = out + OUT_ATTN_OFF;
  float* lOut    = attnOut + ATTN_ELEMS;

  // Workspace layout (f16). Total ~203 MiB. ctx16 aliases x16 (x dead by then).
  _Float16* ws   = (_Float16*)d_ws;
  _Float16* x16  = ws;
  _Float16* wq16 = x16 + (size_t)MM * EE;           // +25165824
  _Float16* wk16 = wq16 + (size_t)EE * EE;          // +589824
  _Float16* wv16 = wk16 + (size_t)EE * EE;
  _Float16* wo16 = wv16 + (size_t)EE * EE;
  _Float16* q16  = wo16 + (size_t)EE * EE;
  _Float16* k16  = q16 + (size_t)MM * EE;
  _Float16* v16  = k16 + (size_t)MM * EE;
  _Float16* p16  = v16 + (size_t)MM * EE;
  _Float16* ctx16 = x16;  // alias

  // 1) f32 -> f16 conversions
  cvt_f32_f16<<<4096, 256, 0, stream>>>(x, x16, MM * EE);
  cvt_f32_f16<<<576, 256, 0, stream>>>(Wq, wq16, EE * EE);
  cvt_f32_f16<<<576, 256, 0, stream>>>(Wk, wk16, EE * EE);
  cvt_f32_f16<<<576, 256, 0, stream>>>(Wv, wv16, EE * EE);
  cvt_f32_f16<<<576, 256, 0, stream>>>(Wo, wo16, EE * EE);

  // 2) QKV projections (M=32768, N=768, K=768), block tile 128x256
  dim3 gproj(EE / 256, MM / 128);
  proj_qkv<0><<<gproj, 256, 0, stream>>>(x16, wq16, bq, q16);
  proj_qkv<1><<<gproj, 256, 0, stream>>>(x16, wk16, bk, k16);
  proj_qkv<2><<<gproj, 256, 0, stream>>>(x16, wv16, bv, v16);

  // 3) Tied-row logits per head (M=N=512, K=4096) -> d_out attn region (f32)
  dim3 glog(CC / 256, CC / 128, HH);
  logits_kernel<<<glog, 256, 0, stream>>>(q16, k16, attnOut);

  // 4) Masked softmax, in-place f32 + f16 copy
  softmax_kernel<<<HH * CC, 256, 0, stream>>>(attnOut, p16);

  // 5) Context per head (M=512, N=4096, K=512)
  dim3 gctx(RD / 256, CC / 128, HH);
  context_kernel<<<gctx, 256, 0, stream>>>(p16, v16, ctx16);

  // 6) Output projection (M=32768, N=768, K=768) -> f32 d_out
  dim3 gout(EE / 256, MM / 128);
  outproj_kernel<<<gout, 256, 0, stream>>>(ctx16, wo16, bo, out);

  // 7) l passthrough
  hipMemcpyAsync(lOut, l, HH * sizeof(float), hipMemcpyDeviceToDevice, stream);
}
